// EBTAttention_54941221651205
// MI455X (gfx1250) — matrix operand order repaired and checked by reading, not compiled, not run on hardware
//
#include <hip/hip_runtime.h>
#include <stdint.h>

typedef __bf16 bf16_t;
typedef __attribute__((ext_vector_type(16))) __bf16 v16bf;
typedef __attribute__((ext_vector_type(8)))  float  v8f;
typedef __attribute__((ext_vector_type(4)))  unsigned int v4u;
typedef __attribute__((ext_vector_type(8)))  int v8i;
typedef __attribute__((ext_vector_type(4)))  int v4i;

#define WMMA_BF16(a, b, c) \
  __builtin_amdgcn_wmma_f32_16x16x32_bf16(false, (a), false, (b), (short)0, (c), false, false)

static constexpr int B_   = 2;
static constexpr int S_   = 2048;
static constexpr int HID_ = 2048;
static constexpr int H_   = 16;
static constexpr int D_   = 128;
static constexpr int NQKV = (H_ + 2) * D_;   // 2304
static constexpr int M_   = B_ * S_;         // 4096

__device__ __forceinline__ uint32_t lds_addr(const void* p) {
  return (uint32_t)(uintptr_t)p;   // flat shared ptr: addr[31:0] == LDS offset
}

// ---------------------------------------------------------------------------
// CDNA5 Tensor Data Mover: one-descriptor 2D tile DMA, global -> LDS.
// D# per CDNA5 ISA (bf16 elements, no pad,
// no iterate, no multicast, no gather). Issue once per wave (EXEC ignored).
// 6-arg builtin form: (v4u g0, v8i g1, v4i g2, v4i g3, v8i extra, i32 cpol).
// ---------------------------------------------------------------------------
__device__ __forceinline__ void tdm_load_2d(uint32_t ldsoff, const void* gptr,
                                            uint32_t tensor_d0, uint32_t tensor_d1,
                                            uint32_t tile_d0,   uint32_t tile_d1,
                                            uint32_t d0_stride)
{
  const uint64_t ga = (uint64_t)(uintptr_t)gptr;
  v4u g0;
  g0[0] = 1u;                                   // count=1, user descriptor
  g0[1] = ldsoff;                               // lds_addr (bytes)
  g0[2] = (uint32_t)(ga & 0xffffffffu);         // global_addr[31:0]
  g0[3] = (uint32_t)((ga >> 32) & 0x01ffffffu)  // global_addr[56:32]
        | (2u << 30);                           // type = 2 ("image")
  v8i g1;
  g1[0] = (int)(1u << 16);                      // workgroup_mask=0, data_size=1 (2B)
  g1[1] = (int)((tensor_d0 & 0xffffu) << 16);   // abar_addr=0 | tensor_dim0[15:0]
  g1[2] = (int)(((tensor_d0 >> 16) & 0xffffu) | ((tensor_d1 & 0xffffu) << 16));
  g1[3] = (int)(((tensor_d1 >> 16) & 0xffffu) | ((tile_d0 & 0xffffu) << 16));
  g1[4] = (int)(tile_d1 & 0xffffu);             // tile_dim1 | tile_dim2=0
  g1[5] = (int)d0_stride;                       // tensor_dim0_stride[31:0]
  g1[6] = 0;                                    // stride[47:32] | dim1_stride[15:0]
  g1[7] = 0;                                    // dim1_stride[47:16]
  const v4i z4 = {0, 0, 0, 0};                  // groups 2/3 unused (2D tensor)
  const v8i z8 = {0, 0, 0, 0, 0, 0, 0, 0};
  __builtin_amdgcn_tensor_load_to_lds(g0, g1, z4, z4, z8, 0);
}

// ---------------------------------------------------------------------------
// 1a. fp32 [K][N] -> bf16 [N][K] transpose+convert (LDS tiled)
// ---------------------------------------------------------------------------
__global__ __launch_bounds__(256)
void transpose_convert_kernel(const float* __restrict__ in, bf16_t* __restrict__ out,
                              int K, int N)
{
  __shared__ float tile[16][17];
  const int k0 = blockIdx.y * 16;
  const int n0 = blockIdx.x * 16;
  const int tx = threadIdx.x, ty = threadIdx.y;
  tile[ty][tx] = in[(size_t)(k0 + ty) * N + (n0 + tx)];
  __syncthreads();
  out[(size_t)(n0 + ty) * K + (k0 + tx)] = (bf16_t)tile[tx][ty];
}

// ---------------------------------------------------------------------------
// 1b. fp32 -> bf16 elementwise convert (8 elements / thread)
// ---------------------------------------------------------------------------
__global__ __launch_bounds__(256)
void convert_bf16_kernel(const float* __restrict__ in, bf16_t* __restrict__ out)
{
  const size_t i = ((size_t)blockIdx.x * 256 + threadIdx.x) * 8;
#pragma unroll
  for (int e = 0; e < 8; ++e) out[i + e] = (bf16_t)in[i + e];
}

// ---------------------------------------------------------------------------
// 2. Tiled WMMA GEMM: C[M][N] fp32 = A[M][K] bf16 * Bt[N][K]^T bf16
//    64x64 tile, BK=32, 128 thr = 4 waves (each 32x32 = 2x2 frags).
//    Double-buffered LDS; TDM stages both tiles one step ahead (wave 0).
// ---------------------------------------------------------------------------
__global__ __launch_bounds__(128)
void gemm_bf16_kernel(const bf16_t* __restrict__ A, const bf16_t* __restrict__ Bt,
                      float* __restrict__ C, int M, int N, int K)
{
  __shared__ bf16_t As[2][64][32];
  __shared__ bf16_t Bs[2][64][32];

  const int tid  = threadIdx.x;
  const int wave = tid >> 5;
  const int lane = tid & 31;
  const int hi   = lane >> 4;
  const int lr   = lane & 15;
  const int m0   = blockIdx.y * 64;
  const int n0   = blockIdx.x * 64;
  const int wm   = (wave >> 1) * 32;
  const int wn   = (wave & 1) * 32;

  v8f acc[2][2];
#pragma unroll
  for (int mi = 0; mi < 2; ++mi)
#pragma unroll
    for (int ni = 0; ni < 2; ++ni)
#pragma unroll
      for (int r = 0; r < 8; ++r) acc[mi][ni][r] = 0.0f;

  // TDM tile descriptors: 64 rows x 32 bf16, row stride K
  auto issue = [&](int k0, int buf) {
    if (wave == 0) {
      tdm_load_2d(lds_addr(&As[buf][0][0]), A  + (size_t)m0 * K + k0,
                  (uint32_t)K, (uint32_t)M, 32u, 64u, (uint32_t)K);
      tdm_load_2d(lds_addr(&Bs[buf][0][0]), Bt + (size_t)n0 * K + k0,
                  (uint32_t)K, (uint32_t)N, 32u, 64u, (uint32_t)K);
    }
  };

  const int T = K >> 5;
  issue(0, 0);
  for (int t = 0; t < T; ++t) {
    const int cur = t & 1;
    if (t + 1 < T) {
      issue((t + 1) << 5, cur ^ 1);
      if (wave == 0) __builtin_amdgcn_s_wait_tensorcnt(2);  // tile t landed
    } else {
      if (wave == 0) __builtin_amdgcn_s_wait_tensorcnt(0);
    }
    __syncthreads();

    v16bf af[2], bfr[2];
#pragma unroll
    for (int mi = 0; mi < 2; ++mi) {
      const int row = wm + mi * 16 + lr;
#pragma unroll
      for (int e = 0; e < 16; ++e) {
        const int kk = ((e < 8) ? e : e + 8) + hi * 8;   // A 16x32 layout
        af[mi][e] = As[cur][row][kk];
      }
    }
#pragma unroll
    for (int ni = 0; ni < 2; ++ni) {
      const int col = wn + ni * 16 + lr;
#pragma unroll
      for (int e = 0; e < 16; ++e)
        bfr[ni][e] = Bs[cur][col][((e < 8) ? e : e + 8) + hi * 8];          // B 32x16 layout
    }
#pragma unroll
    for (int mi = 0; mi < 2; ++mi)
#pragma unroll
      for (int ni = 0; ni < 2; ++ni)
        acc[mi][ni] = WMMA_BF16(af[mi], bfr[ni], acc[mi][ni]);

    __syncthreads();   // all reads of buf[cur] done before it is re-staged
  }

#pragma unroll
  for (int mi = 0; mi < 2; ++mi)
#pragma unroll
    for (int ni = 0; ni < 2; ++ni)
#pragma unroll
      for (int r = 0; r < 8; ++r) {
        const int row = m0 + wm + mi * 16 + r + hi * 8;  // C/D layout
        const int col = n0 + wn + ni * 16 + lr;
        C[(size_t)row * N + col] = acc[mi][ni][r];
      }
}

// ---------------------------------------------------------------------------
// 3. RoPE + repack:  qkv fp32 [M][2304] ->
//      Qb bf16 [B,S,H,D] (rope), Kb bf16 [B,S,D] (rope), Vt bf16 [B,D,S]
// ---------------------------------------------------------------------------
__global__ __launch_bounds__(128)
void rope_pack_kernel(const float* __restrict__ qkv,
                      const float* __restrict__ cosb, const float* __restrict__ sinb,
                      bf16_t* __restrict__ Qb, bf16_t* __restrict__ Kb,
                      bf16_t* __restrict__ Vt)
{
  const int d = threadIdx.x;     // 0..127
  const int s = blockIdx.x;      // 0..2047
  const int b = blockIdx.y;      // 0..1
  const size_t row = (size_t)(b * S_ + s);
  const float* src = qkv + row * NQKV;
  const float c  = cosb[s * D_ + d];
  const float sn = sinb[s * D_ + d];
  const int   dp  = (d < 64) ? d + 64 : d - 64;
  const float sgn = (d < 64) ? -1.0f : 1.0f;
#pragma unroll
  for (int h = 0; h < H_; ++h) {
    const float x = src[h * D_ + d];
    const float r = sgn * src[h * D_ + dp];
    Qb[(row * H_ + h) * D_ + d] = (bf16_t)(x * c + r * sn);
  }
  {
    const float x = src[H_ * D_ + d];
    const float r = sgn * src[H_ * D_ + dp];
    Kb[row * D_ + d] = (bf16_t)(x * c + r * sn);
  }
  Vt[((size_t)(b * D_ + d)) * S_ + s] = (bf16_t)src[(H_ + 1) * D_ + d];
}

// ---------------------------------------------------------------------------
// 4. Flash attention (non-causal, MQA).  128 thr = 4 waves, 64 queries/WG.
//    Double-buffered TDM K/V staging; Q frags resident in VGPRs.
// ---------------------------------------------------------------------------
__global__ __launch_bounds__(128)
void attn_kernel(const bf16_t* __restrict__ Qb, const bf16_t* __restrict__ Kb,
                 const bf16_t* __restrict__ Vt, bf16_t* __restrict__ Ob)
{
  __shared__ bf16_t Ks[2][32][128];   // [buf][key][d]
  __shared__ bf16_t Vs[2][128][32];   // [buf][d][key]
  __shared__ bf16_t Ps[4][16][32];    // per-wave P restripe

  const int tid  = threadIdx.x;
  const int wave = tid >> 5;
  const int lane = tid & 31;
  const int hi   = lane >> 4;
  const int lr   = lane & 15;
  const int b    = blockIdx.z;
  const int h    = blockIdx.y;
  const int q0   = blockIdx.x * 64;
  const float scale = 0.08838834764831845f;  // 1/sqrt(128)

  // Resident Q A-fragments (16 queries x 128 d = 4 frags of 16x32)
  const int q = q0 + wave * 16 + lr;
  v16bf qf[4];
#pragma unroll
  for (int dk = 0; dk < 4; ++dk) {
    const bf16_t* src = Qb + (((size_t)(b * S_ + q) * H_ + h) * D_) + dk * 32 + hi * 8;
#pragma unroll
    for (int e = 0; e < 16; ++e)
      qf[dk][e] = src[(e < 8) ? e : e + 8];
  }

  v8f o[8];
  float m_i[8], l_i[8];
#pragma unroll
  for (int t = 0; t < 8; ++t) {
#pragma unroll
    for (int r = 0; r < 8; ++r) o[t][r] = 0.0f;
    m_i[t] = -__builtin_inff();
    l_i[t] = 0.0f;
  }

  // TDM tile descriptors: K 32x128 from [S][D]; V 128x32 from [D][S]
  auto issueKV = [&](int t0, int buf) {
    if (wave == 0) {
      tdm_load_2d(lds_addr(&Ks[buf][0][0]), Kb + (size_t)(b * S_ + t0) * D_,
                  (uint32_t)D_, (uint32_t)S_, (uint32_t)D_, 32u, (uint32_t)D_);
      tdm_load_2d(lds_addr(&Vs[buf][0][0]), Vt + (size_t)b * D_ * S_ + t0,
                  (uint32_t)S_, (uint32_t)D_, 32u, (uint32_t)D_, (uint32_t)S_);
    }
  };

  const int T = S_ / 32;   // 64
  issueKV(0, 0);
  for (int t = 0; t < T; ++t) {
    const int cur = t & 1;
    if (t + 1 < T) {
      issueKV((t + 1) * 32, cur ^ 1);
      if (wave == 0) __builtin_amdgcn_s_wait_tensorcnt(2);  // tile t landed
    } else {
      if (wave == 0) __builtin_amdgcn_s_wait_tensorcnt(0);
    }
    __syncthreads();

    // scores: two 16x16 tiles, K-dim = 128 in 4 chunks
    v8f c0, c1;
#pragma unroll
    for (int r = 0; r < 8; ++r) { c0[r] = 0.0f; c1[r] = 0.0f; }
#pragma unroll
    for (int dk = 0; dk < 4; ++dk) {
      v16bf bk0, bk1;
#pragma unroll
      for (int e = 0; e < 16; ++e) {
        const int kk = dk * 32 + ((e < 8) ? e : e + 8) + hi * 8;
        bk0[e] = Ks[cur][lr][kk];
        bk1[e] = Ks[cur][16 + lr][kk];
      }
      c0 = WMMA_BF16(qf[dk], bk0, c0);
      c1 = WMMA_BF16(qf[dk], bk1, c1);
    }

    // online softmax; each (r,half) is one query row across 16 lanes
    float alpha[8];
#pragma unroll
    for (int r = 0; r < 8; ++r) {
      const float s0 = c0[r] * scale;
      const float s1 = c1[r] * scale;
      float rm = fmaxf(s0, s1);
#pragma unroll
      for (int mm = 1; mm < 16; mm <<= 1) rm = fmaxf(rm, __shfl_xor(rm, mm, 32));
      const float mn = fmaxf(m_i[r], rm);
      alpha[r] = __expf(m_i[r] - mn);
      m_i[r] = mn;
      const float e0 = __expf(s0 - mn);
      const float e1 = __expf(s1 - mn);
      float rs = e0 + e1;
#pragma unroll
      for (int mm = 1; mm < 16; mm <<= 1) rs += __shfl_xor(rs, mm, 32);
      l_i[r] = l_i[r] * alpha[r] + rs;
      c0[r] = e0; c1[r] = e1;
    }
#pragma unroll
    for (int tt = 0; tt < 8; ++tt)
#pragma unroll
      for (int r = 0; r < 8; ++r) o[tt][r] *= alpha[r];

    // restripe P (C layout -> A layout) through per-wave LDS
#pragma unroll
    for (int r = 0; r < 8; ++r) {
      Ps[wave][r + hi * 8][lr]      = (bf16_t)c0[r];
      Ps[wave][r + hi * 8][16 + lr] = (bf16_t)c1[r];
    }
    asm volatile("s_wait_dscnt 0" ::: "memory");  // intra-wave LDS RAW
    v16bf pa;
#pragma unroll
    for (int e = 0; e < 16; ++e)
      pa[e] = Ps[wave][lr][((e < 8) ? e : e + 8) + hi * 8];

    // O += P (16x32) * V (32x128)
#pragma unroll
    for (int nd = 0; nd < 8; ++nd) {
      v16bf bv;
#pragma unroll
      for (int e = 0; e < 16; ++e) bv[e] = Vs[cur][nd * 16 + lr][((e < 8) ? e : e + 8) + hi * 8];
      o[nd] = WMMA_BF16(pa, bv, o[nd]);
    }

    __syncthreads();   // reads of buf[cur] done before it is re-staged
  }

  // epilogue: normalize, write bf16 [B,S,H*D]
#pragma unroll
  for (int nd = 0; nd < 8; ++nd)
#pragma unroll
    for (int r = 0; r < 8; ++r) {
      const int qq  = q0 + wave * 16 + r + hi * 8;
      const int col = h * D_ + nd * 16 + lr;
      Ob[(size_t)(b * S_ + qq) * (H_ * D_) + col] = (bf16_t)(o[nd][r] / l_i[r]);
    }
}

// ---------------------------------------------------------------------------
// Host launcher
// ---------------------------------------------------------------------------
extern "C" void kernel_launch(void* const* d_in, const int* in_sizes, int n_in,
                              void* d_out, int out_size, void* d_ws, size_t ws_size,
                              hipStream_t stream)
{
  (void)in_sizes; (void)n_in; (void)out_size; (void)ws_size;
  const float* hidden = (const float*)d_in[0];
  const float* cosb   = (const float*)d_in[1];
  const float* sinb   = (const float*)d_in[2];
  const float* w_qkv  = (const float*)d_in[3];
  const float* w_o    = (const float*)d_in[4];
  float* out = (float*)d_out;

  char* ws = (char*)d_ws;
  constexpr size_t SZ_WQKV = (size_t)NQKV * HID_ * 2;   // bf16 [2304][2048]
  constexpr size_t SZ_WO   = (size_t)HID_ * HID_ * 2;   // bf16 [2048][2048]
  constexpr size_t SZ_XB   = (size_t)M_ * HID_ * 2;     // bf16 hidden
  constexpr size_t SZ_QKV  = (size_t)M_ * NQKV * 4;     // fp32
  constexpr size_t SZ_QB   = (size_t)M_ * H_ * D_ * 2;
  constexpr size_t SZ_KB   = (size_t)M_ * D_ * 2;
  constexpr size_t SZ_VT   = (size_t)B_ * D_ * S_ * 2;
  size_t off = 0;
  bf16_t* Wqkv_t = (bf16_t*)(ws + off); off += SZ_WQKV;
  bf16_t* Wo_t   = (bf16_t*)(ws + off); off += SZ_WO;
  bf16_t* Xb     = (bf16_t*)(ws + off); off += SZ_XB;
  float*  qkv    = (float*) (ws + off); off += SZ_QKV;
  bf16_t* Qb     = (bf16_t*)(ws + off); off += SZ_QB;
  bf16_t* Kb     = (bf16_t*)(ws + off); off += SZ_KB;
  bf16_t* Vt     = (bf16_t*)(ws + off); off += SZ_VT;
  bf16_t* Ob     = (bf16_t*)(ws + off);

  dim3 blkT(16, 16);
  transpose_convert_kernel<<<dim3(NQKV / 16, HID_ / 16), blkT, 0, stream>>>(
      w_qkv, Wqkv_t, HID_, NQKV);
  transpose_convert_kernel<<<dim3(HID_ / 16, HID_ / 16), blkT, 0, stream>>>(
      w_o, Wo_t, HID_, HID_);

  convert_bf16_kernel<<<(int)((size_t)M_ * HID_ / 8 / 256), 256, 0, stream>>>(hidden, Xb);

  gemm_bf16_kernel<<<dim3(NQKV / 64, M_ / 64), 128, 0, stream>>>(
      Xb, Wqkv_t, qkv, M_, NQKV, HID_);

  rope_pack_kernel<<<dim3(S_, B_), 128, 0, stream>>>(qkv, cosb, sinb, Qb, Kb, Vt);

  attn_kernel<<<dim3(S_ / 64, H_, B_), 128, 0, stream>>>(Qb, Kb, Vt, Ob);

  gemm_bf16_kernel<<<dim3(HID_ / 64, M_ / 64), 128, 0, stream>>>(
      Ob, Wo_t, out, M_, HID_, HID_);
}
